// MACE_21328807592095
// MI455X (gfx1250) — matrix operand order repaired and checked by reading, not compiled, not run on hardware
//
#include <hip/hip_runtime.h>
#include <hip/hip_bf16.h>
#include <math.h>

#define NNODES 10000
#define NEDGES 160000
#define NGRAPH 16
#define NSPEC  10
#define NF     64
#define NK     16
#define NBR    8
#define RMAXF  5.0f
#define AVGF   16.0f

typedef __bf16 bf16;
typedef __attribute__((ext_vector_type(16))) __bf16 v16bf;
typedef __attribute__((ext_vector_type(8)))  float  v8f;

// Fast silu: v_exp_f32 + v_rcp_f32 (builtin rcp avoids the IEEE divide expansion).
__device__ inline float siluf(float x)  { return x * __builtin_amdgcn_rcpf(1.f + __expf(-x)); }
__device__ inline float dsiluf(float x) {
  float s = __builtin_amdgcn_rcpf(1.f + __expf(-x));
  return s * (1.f + x * (1.f - s));
}

__device__ inline v8f wmma_bf16f32(v16bf a, v16bf b, v8f c) {
  return __builtin_amdgcn_wmma_f32_16x16x32_bf16(false, a, false, b, (short)0, c, false, false);
}

// ---------------------------------------------------------------------------
// Weight prep: f32 64x64 -> bf16 copies (normal + transposed) pre-swizzled into
// WMMA B-fragment order: [frag(8)][lane(32)][e(16)], frag = kkIdx*4 + colTile.
// Contiguous k order used here (A loads match): lane<16 holds K=kk+e, lane>=16 K=kk+16+e,
// column = colTile*16 + (lane&15).
// ---------------------------------------------------------------------------
__global__ void k_conv_mat(const float* __restrict__ W, bf16* __restrict__ outN,
                           bf16* __restrict__ outT)
{
  int i = blockIdx.x * blockDim.x + threadIdx.x;
  if (i >= 64*64) return;
  int e    = i & 15;
  int lane = (i >> 4) & 31;
  int frag = i >> 9;
  int kkIdx = frag >> 2, t = frag & 3;
  bool hi = lane >= 16;
  int k   = kkIdx*32 + (hi ? 16 + e : e);
  int col = t*16 + (lane & 15);
  outN[i] = (bf16)W[k*64 + col];     // B = W
  outT[i] = (bf16)W[col*64 + k];     // B = W^T  (Bt[k][col] = W[col][k])
}

// ---------------------------------------------------------------------------
// Generic C[M,64] = (ACT? silu(A) : A)[M,64] @ B[64,64]   (bf16 WMMA, f32 acc)
// A rows at stride lda (f32); B bf16 in swizzled fragment order; C stride ldc.
// B is staged to LDS with gfx1250 async direct-to-LDS loads (ASYNCcnt).
// ACT/ACCUM are template params -> no per-element uniform branches.
// ---------------------------------------------------------------------------
template <int ACT, int ACCUM>
__global__ __launch_bounds__(256)
void k_gemm_n64(const float* __restrict__ A, int lda,
                const bf16* __restrict__ B,
                float* __restrict__ C, int ldc,
                int M, float scale)
{
  __shared__ __align__(32) bf16 Bs[64*64];         // 8 KB, fragment order
  int tid = threadIdx.x;
  {
    // generic pointer from LDS = {aperture, offset}; low 32 bits = LDS offset
    unsigned ldsbase = (unsigned)(uintptr_t)(void*)Bs;
    const uint4* bsrc = (const uint4*)B;           // 4096 bf16 = 512 * 16B
    for (int i = tid; i < 512; i += 256) {
      unsigned dst = ldsbase + (unsigned)i * 16u;
      const uint4* src = bsrc + i;
      asm volatile("global_load_async_to_lds_b128 %0, %1, off"
                   :: "v"(dst), "v"(src) : "memory");
    }
    asm volatile("s_wait_asynccnt 0" ::: "memory");
  }
  __syncthreads();

  int wave = tid >> 5, lane = tid & 31;
  bool hi = lane >= 16;
  int rowTile = (blockIdx.x * 8 + wave) * 16;
  if (rowTile >= M) return;

  v8f acc[4];
  #pragma unroll
  for (int t = 0; t < 4; ++t)
    #pragma unroll
    for (int i = 0; i < 8; ++i) acc[t][i] = 0.f;

  int mr = rowTile + (lane & 15);
  if (mr >= M) mr = M - 1;                         // duplicate row; store is guarded
  const float* arow = A + (size_t)mr * lda;
  __builtin_prefetch(arow, 0, 1);

  #pragma unroll
  for (int kkIdx = 0; kkIdx < 2; ++kkIdx) {
    int kk = kkIdx * 32;
    // A fragment: lane-lo needs k in [kk,kk+16); hi needs [kk+16,kk+32).
    float av[16];
    const float4* p0 = (const float4*)(arow + kk + (hi ? 16 : 0));
    const float4* p1 = (const float4*)(arow + kk + 8 + (hi ? 16 : 0));
    *(float4*)(av + 0)  = p0[0];
    *(float4*)(av + 4)  = p0[1];
    *(float4*)(av + 8)  = p1[0];
    *(float4*)(av + 12) = p1[1];
    v16bf a;
    #pragma unroll
    for (int e = 0; e < 16; ++e) {
      float x = av[e];
      if (ACT) x = siluf(x);
      a[e] = (bf16)x;
    }
    #pragma unroll
    for (int t = 0; t < 4; ++t) {
      const v16bf* bp = (const v16bf*)(Bs + (size_t)((kkIdx*4 + t)*32 + lane) * 16);
      acc[t] = wmma_bf16f32(a, *bp, acc[t]);
    }
  }

  if (rowTile + 16 <= M) {                         // full tile: no row guards
    #pragma unroll
    for (int t = 0; t < 4; ++t) {
      int col = t*16 + (lane & 15);
      #pragma unroll
      for (int i = 0; i < 8; ++i) {
        int row = rowTile + i + (hi ? 8 : 0);
        float v = acc[t][i] * scale;
        float* p = C + (size_t)row * ldc + col;
        if (ACCUM) *p += v; else *p = v;
      }
    }
  } else {
    #pragma unroll
    for (int t = 0; t < 4; ++t) {
      int col = t*16 + (lane & 15);
      #pragma unroll
      for (int i = 0; i < 8; ++i) {
        int row = rowTile + i + (hi ? 8 : 0);
        if (row < M) {
          float v = acc[t][i] * scale;
          float* p = C + (size_t)row * ldc + col;
          if (ACCUM) *p += v; else *p = v;
        }
      }
    }
  }
}

// ---------------------------------------------------------------------------
// Species-dependent skip: Dst[n] += A[n] (16x64) @ Wsc[species[n]] (64x64).
// One wave per node; B fragments read straight from global (swizzled order).
// ---------------------------------------------------------------------------
__global__ __launch_bounds__(256)
void k_gemm_species(const float* __restrict__ Asrc,   // [N,K,F]
                    const bf16*  __restrict__ Wsc,    // [Z][4096] swizzled bf16
                    const int*   __restrict__ species,
                    float* __restrict__ Dst,          // [N,K,F], accumulated
                    int n_nodes)
{
  int wave = threadIdx.x >> 5, lane = threadIdx.x & 31;
  bool hi = lane >= 16;
  int n = blockIdx.x * 8 + wave;
  if (n >= n_nodes) return;

  const bf16*  Bz   = Wsc + (size_t)species[n] * 64 * 64;
  const float* arow = Asrc + (size_t)n * NK * NF + (size_t)(lane & 15) * NF;
  __builtin_prefetch(arow, 0, 1);

  v8f acc[4];
  #pragma unroll
  for (int t = 0; t < 4; ++t)
    #pragma unroll
    for (int i = 0; i < 8; ++i) acc[t][i] = 0.f;

  #pragma unroll
  for (int kkIdx = 0; kkIdx < 2; ++kkIdx) {
    int kk = kkIdx * 32;
    float av[16];
    const float4* p0 = (const float4*)(arow + kk + (hi ? 16 : 0));
    const float4* p1 = (const float4*)(arow + kk + 8 + (hi ? 16 : 0));
    *(float4*)(av + 0)  = p0[0];
    *(float4*)(av + 4)  = p0[1];
    *(float4*)(av + 8)  = p1[0];
    *(float4*)(av + 12) = p1[1];
    v16bf a;
    #pragma unroll
    for (int e = 0; e < 16; ++e) a[e] = (bf16)av[e];
    #pragma unroll
    for (int t = 0; t < 4; ++t) {
      const v16bf* bp = (const v16bf*)(Bz + (size_t)((kkIdx*4 + t)*32 + lane) * 16);
      acc[t] = wmma_bf16f32(a, *bp, acc[t]);
    }
  }

  float* drow = Dst + (size_t)n * NK * NF;
  #pragma unroll
  for (int t = 0; t < 4; ++t) {
    int col = t*16 + (lane & 15);
    #pragma unroll
    for (int i = 0; i < 8; ++i) {
      int krow = i + (hi ? 8 : 0);
      drow[(size_t)krow * NF + col] += acc[t][i];
    }
  }
}

// ---------------------------------------------------------------------------
// Node species (argmax of one-hot) + E0 segment sum.
// ---------------------------------------------------------------------------
__global__ void k_species_e0(const float* __restrict__ attrs, const float* __restrict__ aE,
                             const int* __restrict__ seg, int* species, float* energyG, int n)
{
  int i = blockIdx.x * blockDim.x + threadIdx.x;
  if (i >= n) return;
  int best = 0; float bv = attrs[(size_t)i * NSPEC];
  #pragma unroll
  for (int z = 1; z < NSPEC; ++z) { float v = attrs[(size_t)i*NSPEC + z]; if (v > bv) { bv = v; best = z; } }
  species[i] = best;
  atomicAdd(&energyG[seg[i]], aE[best]);
}

__global__ void k_embed(const float* __restrict__ Wemb, const int* __restrict__ species,
                        float* __restrict__ nf, int n)
{
  int i = blockIdx.x * blockDim.x + threadIdx.x;
  if (i >= n * NF) return;
  int nn = i >> 6, f = i & 63;
  nf[(size_t)nn * NK * NF + f] = Wemb[species[nn] * NF + f];  // k=0 channel only
}

// ---------------------------------------------------------------------------
// Per-edge geometry: vec, r, Y[16] (float4 stores), ef[8].
// ---------------------------------------------------------------------------
__global__ void k_edge_geom(const float* __restrict__ pos, const float* __restrict__ shifts,
                            const int* __restrict__ snd, const int* __restrict__ rcv,
                            float* vecb, float* rb, float* Yb, float* efb, int E)
{
  int e = blockIdx.x * blockDim.x + threadIdx.x;
  if (e >= E) return;
  int s = snd[e], rC = rcv[e];
  float vx = pos[rC*3+0]-pos[s*3+0]+shifts[e*3+0];
  float vy = pos[rC*3+1]-pos[s*3+1]+shifts[e*3+1];
  float vz = pos[rC*3+2]-pos[s*3+2]+shifts[e*3+2];
  float r = sqrtf(vx*vx + vy*vy + vz*vz);
  float rr = r + 1e-9f;
  vecb[e*3+0]=vx; vecb[e*3+1]=vy; vecb[e*3+2]=vz; rb[e]=r;
  float x = vx/rr, y = vy/rr, z = vz/rr;

  const float s3=1.7320508f, s15=3.8729833f, s5=2.2360680f;
  const float s358=2.0916500f, s105=10.2469508f, s218=1.6201852f, s7=2.6457513f;
  float4* Yv = (float4*)(Yb + (size_t)e*NK);
  Yv[0] = make_float4(1.f, s3*x, s3*y, s3*z);
  Yv[1] = make_float4(s15*x*y, s15*y*z, 0.5f*s5*(3.f*z*z-1.f), s15*x*z);
  Yv[2] = make_float4(0.5f*s15*(x*x-y*y), s358*y*(3.f*x*x-y*y), s105*x*y*z,
                      s218*y*(5.f*z*z-1.f));
  Yv[3] = make_float4(0.5f*s7*z*(5.f*z*z-3.f), s218*x*(5.f*z*z-1.f),
                      0.5f*s105*z*(x*x-y*y), s358*x*(x*x-y*y));

  float xr = r / RMAXF;
  float env = 0.f;
  if (xr < 1.f) {
    float x5 = xr*xr*xr*xr*xr;
    env = 1.f - 21.f*x5 + 35.f*x5*xr - 15.f*x5*xr*xr;   // P=5 polynomial cutoff
  }
  const float c0 = 0.63245553f;                          // sqrt(2/RMAX)
  const float PI = 3.14159265358979323846f;
  #pragma unroll
  for (int nb = 1; nb <= NBR; ++nb)
    efb[(size_t)e*NBR + nb-1] = c0 * sinf(PI*nb*xr)/rr * env;
}

// pre[e,f] = sum_b ef[e,b]*Wr1[b,f]
__global__ void k_radial_pre(const float* __restrict__ efb, const float* __restrict__ Wr1,
                             float* __restrict__ pre, int total)
{
  int i = blockIdx.x * blockDim.x + threadIdx.x;
  if (i >= total) return;
  int e = i >> 6, f = i & 63;
  float acc = 0.f;
  #pragma unroll
  for (int b = 0; b < NBR; ++b) acc += efb[(size_t)e*NBR + b] * Wr1[b*NF + f];
  pre[i] = acc;
}

// Fused edge tensor product + scatter: agg[recv,k,f] += s*rad*Y[k]/AVG
__global__ __launch_bounds__(256)
void k_msg_fwd(const float* __restrict__ s_node, const float* __restrict__ rad,
               const float* __restrict__ Yb, const int* __restrict__ snd,
               const int* __restrict__ rcv, float* __restrict__ agg, int E)
{
  int e = blockIdx.x;
  int tid = threadIdx.x;
  int f = tid & 63, kq = tid >> 6;
  int s = snd[e], rC = rcv[e];
  float t = s_node[(size_t)s*NF + f] * rad[(size_t)e*NF + f] * (1.f/AVGF);
  float4 yv = ((const float4*)(Yb + (size_t)e*NK))[kq];
  float ys[4] = {yv.x, yv.y, yv.z, yv.w};
  float* base = agg + (size_t)rC * NK * NF + (size_t)kq*4*NF + f;
  #pragma unroll
  for (int j = 0; j < 4; ++j)
    atomicAdd(base + (size_t)j*NF, t * ys[j]);
}

// inv, mp = m*poly
__global__ void k_node_poly(const float* __restrict__ m, const float* __restrict__ wpoly,
                            const int* __restrict__ species, float* __restrict__ invb,
                            float* __restrict__ mp, int n)
{
  int i = blockIdx.x * blockDim.x + threadIdx.x;
  if (i >= n * NF) return;
  int nn = i >> 6, f = i & 63;
  const float* mrow = m + (size_t)nn * NK * NF + f;
  float inv = 0.f;
  #pragma unroll
  for (int k = 0; k < NK; ++k) { float v = mrow[(size_t)k*NF]; inv += v*v; }
  invb[i] = inv;
  const float* c = wpoly + ((size_t)species[nn]*NF + f) * 3;
  float poly = c[0] + c[1]*inv + c[2]*inv*inv;
  float* mprow = mp + (size_t)nn * NK * NF + f;
  #pragma unroll
  for (int k = 0; k < NK; ++k) mprow[(size_t)k*NF] = mrow[(size_t)k*NF] * poly;
}

__global__ void k_readout0(const float* __restrict__ nf, const float* __restrict__ w,
                           const int* __restrict__ seg, float* energyG, int n)
{
  int i = blockIdx.x * blockDim.x + threadIdx.x;
  if (i >= n) return;
  const float* row = nf + (size_t)i * NK * NF;
  float acc = 0.f;
  #pragma unroll
  for (int f = 0; f < NF; ++f) acc += row[f] * w[f];
  atomicAdd(&energyG[seg[i]], acc);
}

__global__ void k_readout1(const float* __restrict__ nf, const float* __restrict__ W1,
                           const float* __restrict__ w2, const int* __restrict__ seg,
                           float* __restrict__ hro, float* energyG, int n)
{
  int i = blockIdx.x * blockDim.x + threadIdx.x;
  if (i >= n) return;
  const float* row = nf + (size_t)i * NK * NF;
  float acc = 0.f;
  #pragma unroll
  for (int j = 0; j < 16; ++j) {
    float h = 0.f;
    for (int f = 0; f < NF; ++f) h += row[f] * W1[f*16 + j];
    hro[(size_t)i*16 + j] = h;
    acc += siluf(h) * w2[j];
  }
  atomicAdd(&energyG[seg[i]], acc);
}

// seed: g_nf2[n,0,f] = d(readout1)/d nf
__global__ void k_seed(const float* __restrict__ hro, const float* __restrict__ W1,
                       const float* __restrict__ w2, float* __restrict__ g, int n)
{
  int i = blockIdx.x * blockDim.x + threadIdx.x;
  if (i >= n * NF) return;
  int nn = i >> 6, f = i & 63;
  float acc = 0.f;
  #pragma unroll
  for (int j = 0; j < 16; ++j)
    acc += w2[j] * dsiluf(hro[(size_t)nn*16 + j]) * W1[f*16 + j];
  g[(size_t)nn * NK * NF + f] = acc;
}

__global__ void k_addro0(float* __restrict__ g, const float* __restrict__ w, int n)
{
  int i = blockIdx.x * blockDim.x + threadIdx.x;
  if (i >= n * NF) return;
  int nn = i >> 6, f = i & 63;
  g[(size_t)nn * NK * NF + f] += w[f];
}

// in-place: g_mp -> g_m
__global__ void k_node_poly_bwd(float* __restrict__ gmp, const float* __restrict__ m,
                                const float* __restrict__ invb, const float* __restrict__ wpoly,
                                const int* __restrict__ species, int n)
{
  int i = blockIdx.x * blockDim.x + threadIdx.x;
  if (i >= n * NF) return;
  int nn = i >> 6, f = i & 63;
  const float* c = wpoly + ((size_t)species[nn]*NF + f) * 3;
  float inv = invb[i];
  float poly = c[0] + c[1]*inv + c[2]*inv*inv;
  const float* mrow = m + (size_t)nn * NK * NF + f;
  float* grow = gmp + (size_t)nn * NK * NF + f;
  float gp = 0.f;
  #pragma unroll
  for (int k = 0; k < NK; ++k) gp += grow[(size_t)k*NF] * mrow[(size_t)k*NF];
  float ginv = gp * (c[1] + 2.f*c[2]*inv);
  #pragma unroll
  for (int k = 0; k < NK; ++k)
    grow[(size_t)k*NF] = grow[(size_t)k*NF]*poly + 2.f*mrow[(size_t)k*NF]*ginv;
}

// per-edge backward of tensor product + scatter
__global__ __launch_bounds__(256)
void k_msg_bwd(const float* __restrict__ gagg, const float* __restrict__ s_node,
               const float* __restrict__ rad, const float* __restrict__ Yb,
               const int* __restrict__ snd, const int* __restrict__ rcv,
               float* __restrict__ grad_rad, float* __restrict__ gxu0,
               float* __restrict__ gY, int E)
{
  __shared__ float tsh[64];
  int e = blockIdx.x;
  int tid = threadIdx.x;
  int s = snd[e], rC = rcv[e];
  float sv = 0.f, rv = 0.f;
  if (tid < 64) {
    sv = s_node[(size_t)s*NF + tid];
    rv = rad[(size_t)e*NF + tid];
    tsh[tid] = sv * rv;
  }
  __syncthreads();
  const float* gbase = gagg + (size_t)rC * NK * NF;
  const float ia = 1.f / AVGF;
  if (tid < 64) {
    float gt = 0.f;
    #pragma unroll
    for (int k = 0; k < NK; ++k) gt += gbase[(size_t)k*NF + tid] * Yb[(size_t)e*NK + k];
    gt *= ia;
    grad_rad[(size_t)e*NF + tid] = gt * sv;
    atomicAdd(&gxu0[(size_t)s*NF + tid], gt * rv);
  } else if (tid < 80) {
    int k = tid - 64;
    float gy = 0.f;
    for (int f = 0; f < NF; ++f) gy += gbase[(size_t)k*NF + f] * tsh[f];
    gY[(size_t)e*NK + k] += gy * ia;
  }
}

__global__ void k_silu_bwd(float* __restrict__ g, const float* __restrict__ pre, int total)
{
  int i = blockIdx.x * blockDim.x + threadIdx.x;
  if (i >= total) return;
  g[i] *= dsiluf(pre[i]);
}

// g_ef[e,b] += sum_f g_pre[e,f]*Wr1[b,f]
__global__ void k_gef(const float* __restrict__ gpre, const float* __restrict__ Wr1,
                      float* __restrict__ gef, int total)
{
  int i = blockIdx.x * blockDim.x + threadIdx.x;
  if (i >= total) return;
  int e = i >> 3, b = i & 7;
  const float* gp = gpre + (size_t)e * NF;
  const float* w  = Wr1 + b * NF;
  float acc = 0.f;
  #pragma unroll
  for (int f = 0; f < NF; ++f) acc += gp[f] * w[f];
  gef[i] += acc;
}

// chain g_Y, g_ef back to vec; scatter to grad_pos
__global__ void k_edge_force(const float* __restrict__ gY, const float* __restrict__ gef,
                             const float* __restrict__ vecb, const float* __restrict__ rb,
                             const int* __restrict__ snd, const int* __restrict__ rcv,
                             float* __restrict__ gpos, int E)
{
  int e = blockIdx.x * blockDim.x + threadIdx.x;
  if (e >= E) return;
  float vx = vecb[e*3+0], vy = vecb[e*3+1], vz = vecb[e*3+2];
  float r = rb[e], rr = r + 1e-9f;
  float x = vx/rr, y = vy/rr, z = vz/rr;
  const float* g = gY + (size_t)e * NK;

  const float s3=1.7320508f, s15=3.8729833f, s5=2.2360680f;
  const float s358=2.0916500f, s105=10.2469508f, s218=1.6201852f, s7=2.6457513f;
  float gx=0.f, gyv=0.f, gz=0.f;
  gx  += g[1]*s3;  gyv += g[2]*s3;  gz += g[3]*s3;
  gx  += g[4]*s15*y;             gyv += g[4]*s15*x;
  gyv += g[5]*s15*z;             gz  += g[5]*s15*y;
  gz  += g[6]*3.f*s5*z;
  gx  += g[7]*s15*z;             gz  += g[7]*s15*x;
  gx  += g[8]*s15*x;             gyv -= g[8]*s15*y;
  gx  += g[9]*s358*6.f*x*y;      gyv += g[9]*s358*3.f*(x*x - y*y);
  gx  += g[10]*s105*y*z;         gyv += g[10]*s105*x*z;   gz += g[10]*s105*x*y;
  gyv += g[11]*s218*(5.f*z*z-1.f);                        gz += g[11]*s218*10.f*y*z;
  gz  += g[12]*0.5f*s7*(15.f*z*z-3.f);
  gx  += g[13]*s218*(5.f*z*z-1.f);                        gz += g[13]*s218*10.f*x*z;
  gx  += g[14]*s105*x*z;         gyv -= g[14]*s105*y*z;   gz += g[14]*0.5f*s105*(x*x-y*y);
  gx  += g[15]*s358*(3.f*x*x-y*y); gyv -= g[15]*s358*2.f*x*y;

  // du/dvec chain: gv = (gu - u (u.gu)) / rr
  float udg = x*gx + y*gyv + z*gz;
  float gvx = (gx  - x*udg)/rr;
  float gvy = (gyv - y*udg)/rr;
  float gvz = (gz  - z*udg)/rr;

  // radial chain
  float xr = r / RMAXF;
  float env = 0.f, denv = 0.f;
  if (xr < 1.f) {
    float x4 = xr*xr*xr*xr;
    env  = 1.f - 21.f*x4*xr + 35.f*x4*xr*xr - 15.f*x4*xr*xr*xr;
    denv = (-105.f*x4 + 210.f*x4*xr - 105.f*x4*xr*xr) / RMAXF;
  }
  const float c0 = 0.63245553f;
  const float PI = 3.14159265358979323846f;
  float gr = 0.f;
  #pragma unroll
  for (int nb = 1; nb <= NBR; ++nb) {
    float a  = PI * nb * xr;
    float sn = sinf(a), cs = cosf(a);
    float b  = c0 * sn / rr;
    float db = c0 * ((PI*nb/RMAXF)*cs/rr - sn/(rr*rr));
    gr += gef[(size_t)e*NBR + nb-1] * (db*env + b*denv);
  }
  float ir = 1.f / fmaxf(r, 1e-9f);
  gvx += gr * vx * ir; gvy += gr * vy * ir; gvz += gr * vz * ir;

  int s = snd[e], rC = rcv[e];
  atomicAdd(&gpos[rC*3+0],  gvx); atomicAdd(&gpos[rC*3+1],  gvy); atomicAdd(&gpos[rC*3+2],  gvz);
  atomicAdd(&gpos[s*3+0],  -gvx); atomicAdd(&gpos[s*3+1],  -gvy); atomicAdd(&gpos[s*3+2],  -gvz);
}

__global__ void k_finalize(const float* __restrict__ energyG, const float* __restrict__ gpos,
                           float* __restrict__ out, int total)
{
  int i = blockIdx.x * blockDim.x + threadIdx.x;
  if (i >= total) return;
  out[i] = (i < NGRAPH) ? energyG[i] : -gpos[i - NGRAPH];
}

// ===========================================================================
extern "C" void kernel_launch(void* const* d_in, const int* in_sizes, int n_in,
                              void* d_out, int out_size, void* d_ws, size_t ws_size,
                              hipStream_t stream)
{
  const float* pos    = (const float*)d_in[0];
  const float* attrs  = (const float*)d_in[1];
  const float* shifts = (const float*)d_in[2];
  const float* aE     = (const float*)d_in[3];
  const float* Wemb   = (const float*)d_in[4];
  const float* W_up[2]  = {(const float*)d_in[5],  (const float*)d_in[12]};
  const float* Wr1[2]   = {(const float*)d_in[6],  (const float*)d_in[13]};
  const float* Wr2[2]   = {(const float*)d_in[7],  (const float*)d_in[14]};
  const float* Wmsg[2]  = {(const float*)d_in[8],  (const float*)d_in[15]};
  const float* Wsc[2]   = {(const float*)d_in[9],  (const float*)d_in[16]};
  const float* wpoly[2] = {(const float*)d_in[10], (const float*)d_in[17]};
  const float* Wprod[2] = {(const float*)d_in[11], (const float*)d_in[18]};
  const float* wro0  = (const float*)d_in[19];
  const float* Wro1a = (const float*)d_in[20];
  const float* wro1b = (const float*)d_in[21];
  const int* snd = (const int*)d_in[22];
  const int* rcv = (const int*)d_in[23];
  const int* seg = (const int*)d_in[24];
  float* out = (float*)d_out;

  const int N = NNODES, E = NEDGES;
  const size_t NKF = (size_t)N * NK * NF;

  char* ws = (char*)d_ws;
  size_t off = 0;
  auto alloc = [&](size_t nbytes) { void* p = ws + off; off += (nbytes + 63) & ~(size_t)63; return p; };
  auto allocF = [&](size_t n) { return (float*)alloc(n * sizeof(float)); };
  auto allocB = [&](size_t n) { return (bf16*)alloc(n * sizeof(bf16)); };

  int*   species = (int*)alloc((size_t)N * sizeof(int));
  float* energyG = allocF(NGRAPH);
  float* gpos    = allocF((size_t)N * 3);
  float* vecb    = allocF((size_t)E * 3);
  float* rb      = allocF(E);
  float* Yb      = allocF((size_t)E * NK);
  float* efb     = allocF((size_t)E * NBR);
  float* nfA     = allocF(NKF);
  float* nfB     = allocF(NKF);
  float* agg     = allocF(NKF);       // fwd: agg, bwd: g_agg
  float* mp      = allocF(NKF);       // fwd: m*poly, bwd: g_mp -> g_m
  float* gA      = allocF(NKF);
  float* gB      = allocF(NKF);
  float* gxu0    = allocF((size_t)N * NF);
  float* gradE   = allocF((size_t)E * NF);   // g_rad
  float* ghE     = allocF((size_t)E * NF);   // g_h -> g_pre
  float* gY      = allocF((size_t)E * NK);
  float* gef     = allocF((size_t)E * NBR);
  float* hro     = allocF((size_t)N * 16);
  float* preL[2], *radL[2], *snodeL[2], *mL[2], *invL[2];
  for (int l = 0; l < 2; ++l) {
    preL[l]   = allocF((size_t)E * NF);
    radL[l]   = allocF((size_t)E * NF);
    snodeL[l] = allocF((size_t)N * NF);
    mL[l]     = allocF(NKF);
    invL[l]   = allocF((size_t)N * NF);
  }
  bf16 *upN[2], *upT[2], *r2N[2], *r2T[2], *msgN[2], *msgT[2], *prodN[2], *prodT[2], *scN[2], *scT[2];
  for (int l = 0; l < 2; ++l) {
    upN[l]=allocB(4096);  upT[l]=allocB(4096);
    r2N[l]=allocB(4096);  r2T[l]=allocB(4096);
    msgN[l]=allocB(4096); msgT[l]=allocB(4096);
    prodN[l]=allocB(4096);prodT[l]=allocB(4096);
    scN[l]=allocB((size_t)NSPEC*4096); scT[l]=allocB((size_t)NSPEC*4096);
  }
  (void)ws_size; (void)in_sizes; (void)n_in;

  // --- weight prep (emit fragment-swizzled bf16 copies) ---
  for (int l = 0; l < 2; ++l) {
    k_conv_mat<<<16, 256, 0, stream>>>(W_up[l],  upN[l],  upT[l]);
    k_conv_mat<<<16, 256, 0, stream>>>(Wr2[l],   r2N[l],  r2T[l]);
    k_conv_mat<<<16, 256, 0, stream>>>(Wmsg[l],  msgN[l], msgT[l]);
    k_conv_mat<<<16, 256, 0, stream>>>(Wprod[l], prodN[l],prodT[l]);
    for (int z = 0; z < NSPEC; ++z)
      k_conv_mat<<<16, 256, 0, stream>>>(Wsc[l] + (size_t)z*4096, scN[l] + (size_t)z*4096, scT[l] + (size_t)z*4096);
  }

  // --- init ---
  hipMemsetAsync(energyG, 0, NGRAPH*sizeof(float), stream);
  hipMemsetAsync(gpos,    0, (size_t)N*3*sizeof(float), stream);
  hipMemsetAsync(gY,      0, (size_t)E*NK*sizeof(float), stream);
  hipMemsetAsync(gef,     0, (size_t)E*NBR*sizeof(float), stream);
  hipMemsetAsync(nfA,     0, NKF*sizeof(float), stream);

  k_species_e0<<<(N+255)/256, 256, 0, stream>>>(attrs, aE, seg, species, energyG, N);
  k_edge_geom<<<(E+255)/256, 256, 0, stream>>>(pos, shifts, snd, rcv, vecb, rb, Yb, efb, E);
  k_embed<<<(N*NF+255)/256, 256, 0, stream>>>(Wemb, species, nfA, N);

  const int blkN  = (N + 127) / 128;
  const int blkNK = (N*NK + 127) / 128;
  const int blkE  = (E + 127) / 128;

  // --- forward ---
  float* nf_in = nfA; float* nf_out = nfB;
  for (int l = 0; l < 2; ++l) {
    k_radial_pre<<<(E*NF+255)/256, 256, 0, stream>>>(efb, Wr1[l], preL[l], E*NF);
    k_gemm_n64<1,0><<<blkE, 256, 0, stream>>>(preL[l], NF, r2N[l], radL[l], NF, E, 1.f);
    k_gemm_n64<0,0><<<blkN, 256, 0, stream>>>(nf_in, NK*NF, upN[l], snodeL[l], NF, N, 1.f);
    hipMemsetAsync(agg, 0, NKF*sizeof(float), stream);
    k_msg_fwd<<<E, 256, 0, stream>>>(snodeL[l], radL[l], Yb, snd, rcv, agg, E);
    k_gemm_n64<0,0><<<blkNK, 256, 0, stream>>>(agg, NF, msgN[l], mL[l], NF, N*NK, 1.f);
    k_node_poly<<<(N*NF+255)/256, 256, 0, stream>>>(mL[l], wpoly[l], species, invL[l], mp, N);
    k_gemm_n64<0,0><<<blkNK, 256, 0, stream>>>(mp, NF, prodN[l], nf_out, NF, N*NK, 1.f);
    k_gemm_species<<<(N+7)/8, 256, 0, stream>>>(nf_in, scN[l], species, nf_out, N);
    if (l == 0) k_readout0<<<(N+255)/256, 256, 0, stream>>>(nf_out, wro0, seg, energyG, N);
    else        k_readout1<<<(N+255)/256, 256, 0, stream>>>(nf_out, Wro1a, wro1b, seg, hro, energyG, N);
    float* tmp = nf_in; nf_in = nf_out; nf_out = tmp;
  }

  // --- backward ---
  auto bwd_layer = [&](int l, float* gout, float* gin, bool need_gin) {
    k_gemm_n64<0,0><<<blkNK, 256, 0, stream>>>(gout, NF, prodT[l], mp, NF, N*NK, 1.f);
    k_node_poly_bwd<<<(N*NF+255)/256, 256, 0, stream>>>(mp, mL[l], invL[l], wpoly[l], species, N);
    k_gemm_n64<0,0><<<blkNK, 256, 0, stream>>>(mp, NF, msgT[l], agg, NF, N*NK, 1.f);
    hipMemsetAsync(gxu0, 0, (size_t)N*NF*sizeof(float), stream);
    k_msg_bwd<<<E, 256, 0, stream>>>(agg, snodeL[l], radL[l], Yb, snd, rcv, gradE, gxu0, gY, E);
    k_gemm_n64<0,0><<<blkE, 256, 0, stream>>>(gradE, NF, r2T[l], ghE, NF, E, 1.f);
    k_silu_bwd<<<(E*NF+255)/256, 256, 0, stream>>>(ghE, preL[l], E*NF);
    k_gef<<<(E*NBR+255)/256, 256, 0, stream>>>(ghE, Wr1[l], gef, E*NBR);
    if (need_gin) {
      hipMemsetAsync(gin, 0, NKF*sizeof(float), stream);
      k_gemm_species<<<(N+7)/8, 256, 0, stream>>>(gout, scT[l], species, gin, N);
      k_gemm_n64<0,1><<<blkN, 256, 0, stream>>>(gxu0, NF, upT[l], gin, NK*NF, N, 1.f);
    }
  };

  hipMemsetAsync(gA, 0, NKF*sizeof(float), stream);
  k_seed<<<(N*NF+255)/256, 256, 0, stream>>>(hro, Wro1a, wro1b, gA, N);
  bwd_layer(1, gA, gB, true);
  k_addro0<<<(N*NF+255)/256, 256, 0, stream>>>(gB, wro0, N);
  bwd_layer(0, gB, gA, false);   // nf0 does not depend on positions

  k_edge_force<<<(E+255)/256, 256, 0, stream>>>(gY, gef, vecb, rb, snd, rcv, gpos, E);
  k_finalize<<<(out_size+255)/256, 256, 0, stream>>>(energyG, gpos, out, out_size);
}
